// HdcCNNEncoder_69535520522622
// MI455X (gfx1250) — matrix operand order repaired and checked by reading, not compiled, not run on hardware
//
#include <hip/hip_runtime.h>
#include <hip/hip_bf16.h>
#include <stddef.h>

// ---------- problem constants ----------
#define T_LEN  8192
#define L1     8183            // conv1 valid output length
#define L2C    8174            // conv2 valid output length
#define LPOOL  4087            // after MaxPool1d(2)
#define FLATN  (128 * LPOOL)   // 523136
#define DHV    10000
#define KDIM   640             // im2col K, reordered as K' = kx*64 + ci
#define CHUNK  (FLATN / 16)    // 32696, divisible by 4

typedef __attribute__((ext_vector_type(16))) _Float16 v16h;
typedef __attribute__((ext_vector_type(2)))  _Float16 v2h;
typedef __attribute__((ext_vector_type(8)))  float    v8f;
typedef __attribute__((ext_vector_type(4)))  float    v4f;

// ---------- kernel 0: conv2 weights fp32 -> f16, repacked [128][K'=kx*64+ci] ----------
__global__ void castw2_kernel(const float* __restrict__ w, _Float16* __restrict__ wh) {
    int i = blockIdx.x * blockDim.x + threadIdx.x;
    if (i >= 128 * KDIM) return;
    int co = i / KDIM;
    int r  = i - co * KDIM;
    int kx = r >> 6;          // 0..9
    int ci = r & 63;          // 0..63
    wh[i] = (_Float16)w[co * KDIM + ci * 10 + kx];
}

// ---------- kernel 1: conv1 + ReLU, output fp32 [64][L1] ----------
__global__ void conv1_kernel(const float* __restrict__ sig,
                             const float* __restrict__ w1,
                             const float* __restrict__ b1,
                             float* __restrict__ a1f) {
    __shared__ float wrow[40];
    const int c   = blockIdx.y;
    const int tid = threadIdx.x;
    if (tid < 40) wrow[tid] = w1[c * 40 + tid];
    __syncthreads();
    const int t = blockIdx.x * blockDim.x + tid;
    if (t >= L1) return;
    float acc = b1[c];
#pragma unroll
    for (int k = 0; k < 10; ++k) {
        v4f s = *(const v4f*)(sig + (size_t)(t + k) * 4);   // signals [T][4], 16B rows
        acc += wrow[0 * 10 + k] * s.x + wrow[1 * 10 + k] * s.y +
               wrow[2 * 10 + k] * s.z + wrow[3 * 10 + k] * s.w;
    }
    a1f[(size_t)c * L1 + t] = fmaxf(acc, 0.0f);
}

// ---------- kernel 2: conv2 as WMMA GEMM (M=128, K=640, N=8174) ----------
// 8 waves/block; wave w owns out-channels [16w,16w+16); block covers 64 t values.
// K-chunk kc: conv tap kx = kc>>1 (uniform), ci = (kc&1)*32 + kk  -> no division,
// contiguous A1 reads in the staging loop.
#define BS_STRIDE 48   // f16 per LDS row -> 96B rows, keeps v16h loads 32B aligned
__global__ void __launch_bounds__(256)
conv2_wmma_kernel(const float* __restrict__ a1f,
                  const _Float16* __restrict__ w2h,
                  const float* __restrict__ b2,
                  float* __restrict__ O) {
    __shared__ _Float16 Bs[64 * BS_STRIDE];   // [tt][kk], transposed im2col tile
    const int tid  = threadIdx.x;
    const int lane = tid & 31;
    const int wave = tid >> 5;        // 0..7 -> M tile
    const int hlf  = lane >> 4;       // 0: lanes 0-15, 1: lanes 16-31
    const int l15  = lane & 15;
    const int t0   = blockIdx.x * 64;

    // staging role (constant over kc): this thread copies 8 contiguous t values
    const int kkme  = tid >> 3;       // 0..31 (== ci offset within chunk)
    const int tsme  = (tid & 7) * 8;  // 0,8,...,56 (tt base)
    const bool interior = (t0 + 73 <= L1);   // whole staging window in-bounds

    v8f acc[4];
#pragma unroll
    for (int s = 0; s < 4; ++s)
#pragma unroll
        for (int v = 0; v < 8; ++v) acc[s][v] = 0.0f;

    for (int kc = 0; kc < 20; ++kc) {   // K chunks of 32
        const int kx = kc >> 1;                  // conv tap (uniform in chunk)
        const int ci = ((kc & 1) << 5) + kkme;   // input channel for my kk
        const float* src = a1f + (size_t)ci * L1 + (t0 + tsme + kx);

        __syncthreads();
        if (interior) {
#pragma unroll
            for (int e = 0; e < 8; ++e)
                Bs[(tsme + e) * BS_STRIDE + kkme] = (_Float16)src[e];
        } else {
#pragma unroll
            for (int e = 0; e < 8; ++e) {
                int t = t0 + tsme + kx + e;
                float v = (t < L1) ? src[e] : 0.0f;
                Bs[(tsme + e) * BS_STRIDE + kkme] = (_Float16)v;
            }
        }
        __syncthreads();

        // A fragment: lane holds row m = 16*wave + l15; K = 16*hlf .. 16*hlf+15 contiguous
        const int m = wave * 16 + l15;
        v16h afrag;
#pragma unroll
        for (int p = 0; p < 8; ++p) {
            int kb = kc * 32 + 16 * hlf + 2 * p;
            v2h pr = *(const v2h*)(w2h + (size_t)m * KDIM + kb);
            afrag[2 * p]     = pr.x;
            afrag[2 * p + 1] = pr.y;
        }

#pragma unroll
        for (int s = 0; s < 4; ++s) {
            const int tt = s * 16 + l15;   // N index for this lane
            // lane reads K = 16*hlf .. 16*hlf+15 contiguously (2x ds_load_b128)
            v16h bfrag = *reinterpret_cast<const v16h*>(&Bs[tt * BS_STRIDE + 16 * hlf]);
            acc[s] = __builtin_amdgcn_wmma_f32_16x16x32_f16(
                false, afrag, false, bfrag, (short)0, acc[s], false, false);
        }
    }

    // epilogue: bias + ReLU + store O[m][t]
#pragma unroll
    for (int s = 0; s < 4; ++s) {
        const int t = t0 + s * 16 + l15;
        if (t < L2C) {
#pragma unroll
            for (int v = 0; v < 8; ++v) {
                int m = wave * 16 + hlf * 8 + v;
                float val = acc[s][v] + b2[m];
                O[(size_t)m * L2C + t] = fmaxf(val, 0.0f);
            }
        }
    }
}

// ---------- kernel 3: MaxPool1d(2) + flatten -> xf[FLATN] ----------
__global__ void pool_kernel(const float* __restrict__ O, float* __restrict__ xf) {
    int i = blockIdx.x * blockDim.x + threadIdx.x;
    if (i >= FLATN) return;
    int co = i / LPOOL;
    int tp = i - co * LPOOL;
    const float* p = O + (size_t)co * L2C + 2 * tp;
    xf[i] = fmaxf(p[0], p[1]);
}

// ---------- kernel 4: y := lin_b (fresh every call -> graph-replay safe) ----------
__global__ void inity_kernel(const float* __restrict__ lin_b, float* __restrict__ y) {
    int i = threadIdx.x;
    if (i < 128) y[i] = lin_b[i];
}

// ---------- kernel 5: bandwidth-bound GEMV y[r] += lin_w[r,:] . xf ----------
// 268 MB of lin_w streamed once with non-temporal float4 loads (HBM-bound).
__global__ void __launch_bounds__(256)
gemv_kernel(const float* __restrict__ lin_w, const float* __restrict__ xf,
            float* __restrict__ y) {
    const int r     = blockIdx.x >> 4;   // 128 rows
    const int chunk = blockIdx.x & 15;   // 16 chunks per row
    const float* wrow = lin_w + (size_t)r * FLATN + (size_t)chunk * CHUNK;
    const float* xp   = xf + (size_t)chunk * CHUNK;
    float sum = 0.0f;
    const int n4 = CHUNK / 4;            // 8174 float4s
    for (int i = threadIdx.x; i < n4; i += 256) {
        v4f w = __builtin_nontemporal_load((const v4f*)wrow + i);
        v4f x = *((const v4f*)xp + i);
        sum += w.x * x.x + w.y * x.y + w.z * x.z + w.w * x.w;
    }
    __shared__ float red[256];
    red[threadIdx.x] = sum;
    __syncthreads();
    for (int off = 128; off > 0; off >>= 1) {
        if (threadIdx.x < off) red[threadIdx.x] += red[threadIdx.x + off];
        __syncthreads();
    }
    if (threadIdx.x == 0) atomicAdd(&y[r], red[0]);
}

// ---------- kernel 6: HDC sinusoid + feature hypervectors + sign ----------
__global__ void __launch_bounds__(256)
hdc_kernel(const float* __restrict__ y, const float* __restrict__ hdc_w,
           const float* __restrict__ hdc_b, const float* __restrict__ feat,
           const int* __restrict__ fidx, const float* __restrict__ feat_w,
           const float* __restrict__ feat_b, float* __restrict__ out) {
    __shared__ float ry[128];
    __shared__ float fv[18];
    const int tid = threadIdx.x;
    if (tid < 128) ry[tid] = fmaxf(y[tid], 0.0f);    // Linear ReLU applied here
    if (tid < 18)  fv[tid] = feat[fidx[tid]];
    __syncthreads();
    const int d = blockIdx.x * blockDim.x + tid;
    if (d >= DHV) return;

    float proj = 0.0f;
    const float* wrow = hdc_w + (size_t)d * 128;
#pragma unroll 8
    for (int j = 0; j < 128; ++j) proj += ry[j] * wrow[j];
    float sample = cosf(proj + hdc_b[d]) * sinf(proj);

    float h[18];
#pragma unroll
    for (int i = 0; i < 18; ++i) {
        float fp = fv[i] * feat_w[(size_t)i * DHV + d];
        h[i] = cosf(fp + feat_b[(size_t)i * DHV + d]) * sinf(fp);
    }
    float fh = (h[14] + h[11]) * h[16]
             * (h[4] + h[8] + h[6] + h[1] + h[5] + h[12] + h[17])
             * h[13] * (h[15] + h[2]) * h[3]
             * h[0] * h[10] * h[7] * h[9];

    out[d] = (sample + fh) > 0.0f ? 1.0f : -1.0f;
}

// ---------- host launcher ----------
extern "C" void kernel_launch(void* const* d_in, const int* in_sizes, int n_in,
                              void* d_out, int out_size, void* d_ws, size_t ws_size,
                              hipStream_t stream) {
    (void)in_sizes; (void)n_in; (void)out_size; (void)ws_size;

    const float* signals = (const float*)d_in[0];
    const float* feat    = (const float*)d_in[1];
    const float* conv1_w = (const float*)d_in[2];
    const float* conv1_b = (const float*)d_in[3];
    const float* conv2_w = (const float*)d_in[4];
    const float* conv2_b = (const float*)d_in[5];
    const float* lin_w   = (const float*)d_in[6];
    const float* lin_b   = (const float*)d_in[7];
    const float* hdc_w   = (const float*)d_in[8];
    const float* hdc_b   = (const float*)d_in[9];
    const float* feat_w  = (const float*)d_in[10];
    const float* feat_b  = (const float*)d_in[11];
    const int*   feat_ix = (const int*)d_in[12];
    float* out = (float*)d_out;

    // carve scratch from d_ws (aligned to 256B)
    char*  ws  = (char*)d_ws;
    size_t off = 0;
    auto carve = [&](size_t bytes) -> void* {
        void* p = ws + off;
        off = (off + bytes + 255) & ~(size_t)255;
        return p;
    };
    _Float16* w2h = (_Float16*)carve((size_t)128 * KDIM * sizeof(_Float16)); // 160 KB
    float*    a1f = (float*)   carve((size_t)64 * L1 * sizeof(float));       // ~2.1 MB
    float*    O   = (float*)   carve((size_t)128 * L2C * sizeof(float));     // ~4.2 MB
    float*    xf  = (float*)   carve((size_t)FLATN * sizeof(float));         // ~2.1 MB
    float*    yv  = (float*)   carve(128 * sizeof(float));

    castw2_kernel<<<(128 * KDIM + 255) / 256, 256, 0, stream>>>(conv2_w, w2h);
    conv1_kernel<<<dim3((L1 + 255) / 256, 64), 256, 0, stream>>>(signals, conv1_w, conv1_b, a1f);
    conv2_wmma_kernel<<<(L2C + 63) / 64, 256, 0, stream>>>(a1f, w2h, conv2_b, O);
    pool_kernel<<<(FLATN + 255) / 256, 256, 0, stream>>>(O, xf);
    inity_kernel<<<1, 128, 0, stream>>>(lin_b, yv);
    gemv_kernel<<<128 * 16, 256, 0, stream>>>(lin_w, xf, yv);
    hdc_kernel<<<(DHV + 255) / 256, 256, 0, stream>>>(yv, hdc_w, hdc_b, feat,
                                                      feat_ix, feat_w, feat_b, out);
}
